// GaussianGraph_67276367724846
// MI455X (gfx1250) — matrix operand order repaired and checked by reading, not compiled, not run on hardware
//
#include <hip/hip_runtime.h>
#include <hip/hip_bf16.h>
#include <math.h>

typedef _Float16 v16h __attribute__((ext_vector_type(16)));
typedef _Float16 v8h  __attribute__((ext_vector_type(8)));
typedef float    v8f  __attribute__((ext_vector_type(8)));

#define BB 2
#define VV 4
#define HH 256
#define WW 256
#define CCH 64
#define HWHW (HH*WW)
#define GAMMA_C 0.1f

// workspace layout (bytes)
#define OFF_COUNTS 0                       // 16 ints
#define OFF_W2C    256                     // 8 * 12 floats
#define OFF_WP1    1024                    // 36864 f16 = 73728 B
#define OFF_WP2    (1024 + 73728)
#define OFF_FUSED  (1024 + 2*73728)        // 8*65536*64 f16 = 64 MB
#define OFF_Y1     (OFF_FUSED + (size_t)8*HWHW*CCH*2)

// ---------------- init: invert extrinsics (4x4 Gauss-Jordan), zero counters ----
__global__ void k_init(const float* __restrict__ extr, float* __restrict__ w2c,
                       int* __restrict__ counts) {
  int t = threadIdx.x;
  if (t >= 8 && t < 24) counts[t - 8] = 0;
  if (t < 8) {
    float a[4][8];
    const float* m = extr + t * 16;
    for (int r = 0; r < 4; r++)
      for (int c = 0; c < 4; c++) { a[r][c] = m[r*4 + c]; a[r][c+4] = (r == c) ? 1.f : 0.f; }
    for (int col = 0; col < 4; col++) {
      int piv = col; float best = fabsf(a[col][col]);
      for (int r = col + 1; r < 4; r++) { float v = fabsf(a[r][col]); if (v > best) { best = v; piv = r; } }
      if (piv != col) for (int c = 0; c < 8; c++) { float tmp = a[col][c]; a[col][c] = a[piv][c]; a[piv][c] = tmp; }
      float d = 1.f / a[col][col];
      for (int c = 0; c < 8; c++) a[col][c] *= d;
      for (int r = 0; r < 4; r++) if (r != col) {
        float f = a[r][col];
        for (int c = 0; c < 8; c++) a[r][c] -= f * a[col][c];
      }
    }
    float* o = w2c + t * 12;
    for (int r = 0; r < 3; r++)
      for (int c = 0; c < 4; c++) o[r*4 + c] = a[r][c + 4];
  }
}

// ---------------- pack weights into per-lane B-fragment layout (f16) ----------
// packed[kk][nf][lane][e] = B[k = kk*32 + 16*(lane>>4) + e][n = nf*16 + (lane&15)]
// B[k][n] = w[co=n][ci=k%64][ky=(k/64)/3][kx=(k/64)%3]   (OIHW)
__global__ void k_packw(const float* __restrict__ w1, const float* __restrict__ w2,
                        _Float16* __restrict__ wp1, _Float16* __restrict__ wp2) {
  int idx = blockIdx.x * 256 + threadIdx.x;     // 0..73727
  int which = idx / 36864;
  int r = idx - which * 36864;
  int e    = r & 15;
  int lane = (r >> 4) & 31;
  int nf   = (r >> 9) & 3;
  int kk   = r >> 11;
  int k = kk * 32 + ((lane >> 4) << 4) + e;
  int n = nf * 16 + (lane & 15);
  int tap = k >> 6; int ci = k & 63;
  int ky = tap / 3, kx = tap - ky * 3;
  const float* w = which ? w2 : w1;
  float val = w[((n * 64 + ci) * 3 + ky) * 3 + kx];
  (which ? wp2 : wp1)[r] = (_Float16)val;
}

// ---------------- projection helper -------------------------------------------
__device__ __forceinline__ bool project_mask(const float* __restrict__ w2c,
                                             const float* __restrict__ intr,
                                             float px, float py, float pz, int& outIdx) {
  float cx = w2c[0]*px + w2c[1]*py + w2c[2]*pz  + w2c[3];
  float cy = w2c[4]*px + w2c[5]*py + w2c[6]*pz  + w2c[7];
  float cz = w2c[8]*px + w2c[9]*py + w2c[10]*pz + w2c[11];
  bool vz = cz > 1e-8f;
  float inv = 1.f / (cz + 1e-8f);
  float hx = cx * inv, hy = cy * inv, hz = cz * inv;
  float nx = intr[0]*hx + intr[1]*hy + intr[2]*hz;
  float ny = intr[3]*hx + intr[4]*hy + intr[5]*hz;
  bool m = (nx >= 0.f) && (nx < 1.f) && (ny >= 0.f) && (ny < 1.f) && vz;
  int ix = (int)floorf(nx * (float)WW);
  int iy = (int)floorf(ny * (float)HH);
  ix = min(max(ix, 0), WW - 1);
  iy = min(max(iy, 0), HH - 1);
  outIdx = iy * WW + ix;
  return m;
}

// ---------------- branchless exact-grade GELU ---------------------------------
// erf via Abramowitz-Stegun 7.1.26 (|err| <= 1.5e-7), single v_exp, no branches
__device__ __forceinline__ float gelu_exact(float x) {
  float z  = x * 0.70710678118654752f;
  float az = fabsf(z);
  float t  = 1.f / (1.f + 0.3275911f * az);
  float poly = t * (0.254829592f +
               t * (-0.284496736f +
               t * (1.421413741f +
               t * (-1.453152027f +
               t * 1.061405429f))));
  float e = __builtin_amdgcn_exp2f(-az * az * 1.4426950408889634f);
  float erf_z = copysignf(1.f - poly * e, z);
  return 0.5f * x * (1.f + erf_z);
}

// ---------------- global mask counts per (b, j, slot) --------------------------
__global__ void k_count(const float* __restrict__ means, const float* __restrict__ intr_all,
                        const float* __restrict__ w2c_all, int* __restrict__ counts) {
  int pairIdx = blockIdx.y;            // (b*V + j)*2 + slot
  int slot = pairIdx & 1;
  int bj = pairIdx >> 1;
  int j = bj & 3;
  int k = j + (slot ? 1 : -1);
  if (k < 0 || k >= VV) return;
  int b = bj >> 2;
  __shared__ int scount;
  if (threadIdx.x == 0) scount = 0;
  __syncthreads();
  const float* w2c  = w2c_all  + (b * VV + k) * 12;
  const float* intr = intr_all + (b * VV + k) * 9;
  int base = blockIdx.x * 1024;
  int local = 0;
  for (int s = 0; s < 4; s++) {
    int i = base + s * 256 + threadIdx.x;
    const float* p = means + ((size_t)bj * HWHW + i) * 3;
    int gidx;
    if (project_mask(w2c, intr, p[0], p[1], p[2], gidx)) local++;
  }
  atomicAdd(&scount, local);
  __syncthreads();
  if (threadIdx.x == 0) atomicAdd(&counts[pairIdx], scount);
}

// ---------------- fuse: gather + normalize, emit f16 NHWC ----------------------
__global__ void k_fuse(const float* __restrict__ means, const float* __restrict__ feats,
                       const float* __restrict__ intr_all, const float* __restrict__ w2c_all,
                       const int* __restrict__ counts, _Float16* __restrict__ fused) {
  int idx = blockIdx.x * 256 + threadIdx.x;    // B*V*HW*4 threads
  int cg = idx & 3;
  int pt = idx >> 2;
  int i  = pt & (HWHW - 1);
  int bj = pt >> 16;                           // HW = 65536
  int j = bj & 3; int b = bj >> 2;
  int c0 = cg * 16;
  float acc[16];
  const float* f0 = feats + ((size_t)bj * HWHW + i) * CCH + c0;
  #pragma unroll
  for (int c = 0; c < 16; c++) acc[c] = f0[c];
  float norm = 1.f;
  const float* p = means + ((size_t)bj * HWHW + i) * 3;
  float px = p[0], py = p[1], pz = p[2];
  #pragma unroll
  for (int slot = 0; slot < 2; slot++) {
    int k = j + (slot ? 1 : -1);
    if (k < 0 || k >= VV) continue;
    float scale = GAMMA_C * (float)counts[bj * 2 + slot] / (float)HWHW;
    norm += scale;
    const float* w2c  = w2c_all  + (b * VV + k) * 12;
    const float* intr = intr_all + (b * VV + k) * 9;
    int gidx;
    if (project_mask(w2c, intr, px, py, pz, gidx)) {
      const float* g = feats + ((size_t)(b * VV + k) * HWHW + gidx) * CCH + c0;
      #pragma unroll
      for (int c = 0; c < 16; c++) acc[c] += g[c] * scale;
    }
  }
  float inv = 1.f / norm;
  _Float16* o = fused + ((size_t)bj * HWHW + i) * CCH + c0;
  #pragma unroll
  for (int c = 0; c < 16; c++) o[c] = (_Float16)(acc[c] * inv);
}

// ---------------- conv3x3 64->64 as implicit GEMM on WMMA ----------------------
// 128 threads = 4 waves; wave tile M=64 x N=64 (16 accumulators) -> 16 ds_b128
// per 16 WMMA per K-chunk (512 B per WMMA of LDS traffic).
// Epilogue: transpose through LDS (reusing weight buffer) for coalesced
// 128B-per-row global stores instead of scattered b16/b32 stores.
// MODE 0: out = f16 GELU(conv(x)+b)   MODE 1: out = f32 conv(x)+b
template <int MODE>
__global__ __launch_bounds__(128) void k_conv(const _Float16* __restrict__ x,
                                              const _Float16* __restrict__ wpack,
                                              const float* __restrict__ bias,
                                              void* __restrict__ out) {
  __shared__ __align__(16) char sWbuf[73728];         // 72 KB: weights, then out tile
  __shared__ __align__(16) _Float16 sX[324 * CCH];    // 18x18 patch, 40.5 KB
  _Float16* sW = (_Float16*)sWbuf;
  int n   = blockIdx.z;
  int ty0 = blockIdx.y * 16;
  int tx0 = blockIdx.x * 16;
  int tid = threadIdx.x;

  // stage weights (uint4 = 8 halves)
  {
    const uint4* src = (const uint4*)wpack;
    uint4* dst = (uint4*)sW;
    for (int t = tid; t < 36864 / 8; t += 128) dst[t] = src[t];
  }
  // stage input patch with halo, zero-pad OOB
  for (int t = tid; t < 324 * 8; t += 128) {
    int pix = t >> 3, part = t & 7;
    int py = pix / 18, px2 = pix - py * 18;
    int gy = ty0 - 1 + py, gx = tx0 - 1 + px2;
    uint4 v = make_uint4(0u, 0u, 0u, 0u);
    if (gy >= 0 && gy < HH && gx >= 0 && gx < WW)
      v = *(const uint4*)(x + (((size_t)n * HH + gy) * WW + gx) * CCH + part * 8);
    *(uint4*)(sX + pix * CCH + part * 8) = v;
  }
  __syncthreads();

  int lane  = tid & 31;
  int wv    = tid >> 5;       // wave 0..3, owns output pixels wv*64 .. wv*64+63
  int lhalf = lane >> 4;      // 0/1
  int l15   = lane & 15;

  v8f acc[4][4];
  #pragma unroll
  for (int mf = 0; mf < 4; mf++)
    #pragma unroll
    for (int nf = 0; nf < 4; nf++)
      #pragma unroll
      for (int q = 0; q < 8; q++) acc[mf][nf][q] = 0.f;

  for (int kk = 0; kk < 18; kk++) {
    int tap = kk >> 1;
    int ci0 = (kk & 1) << 5;
    int ky = tap / 3, kx = tap - ky * 3;
    int chbase = ci0 + (lhalf << 4);
    // load 4 B fragments (resident for this K-chunk)
    v16h Bf[4];
    #pragma unroll
    for (int nf = 0; nf < 4; nf++)
      Bf[nf] = *(const v16h*)(sW + (((kk * 4 + nf) * 32 + lane) << 4));
    // stream 4 A fragments, 4 WMMA each
    #pragma unroll
    for (int mf = 0; mf < 4; mf++) {
      int m = wv * 64 + mf * 16 + l15;   // A row = lane&15
      int oy = m >> 4, ox = m & 15;
      const _Float16* pa = sX + ((oy + ky) * 18 + (ox + kx)) * CCH + chbase;
      union { v16h v; v8h h[2]; } u;
      u.h[0] = *(const v8h*)pa;          // K chunk {0..7}+16*lhalf
      u.h[1] = *(const v8h*)(pa + 8);    // K chunk {8..15}+16*lhalf
      #pragma unroll
      for (int nf = 0; nf < 4; nf++) {
        acc[mf][nf] = __builtin_amdgcn_wmma_f32_16x16x32_f16(
            false, u.v, false, Bf[nf], (short)0, acc[mf][nf], false, false);
      }
    }
  }

  // ---- epilogue: bias (+GELU), transpose via LDS, coalesced global stores ----
  __syncthreads();   // all waves done reading sW (weights) and sX
  // C/D layout: VGPR r holds M = r + 8*(lane>>4), N = lane&15
  if (MODE == 0) {
    _Float16* sOut = (_Float16*)sWbuf;              // 256 px * 64 ch f16 = 32 KB
    #pragma unroll
    for (int nf = 0; nf < 4; nf++) {
      int ch = nf * 16 + l15;
      float bv = bias[ch];
      #pragma unroll
      for (int mf = 0; mf < 4; mf++) {
        int mbase = wv * 64 + mf * 16 + (lhalf << 3);
        #pragma unroll
        for (int r = 0; r < 8; r++)
          sOut[(mbase + r) * CCH + ch] = (_Float16)gelu_exact(acc[mf][nf][r] + bv);
      }
    }
    __syncthreads();
    _Float16* op = (_Float16*)out;
    for (int t = tid; t < 256 * 8; t += 128) {      // 8 uint4 per pixel
      int pix = t >> 3, part = t & 7;
      int oy = pix >> 4, ox = pix & 15;
      *(uint4*)(op + (((size_t)n * HH + (ty0 + oy)) * WW + (tx0 + ox)) * CCH + part * 8)
          = *(const uint4*)(sOut + pix * CCH + part * 8);
    }
  } else {
    float* sOut = (float*)sWbuf;                    // 256 px * 64 ch f32 = 64 KB
    #pragma unroll
    for (int nf = 0; nf < 4; nf++) {
      int ch = nf * 16 + l15;
      float bv = bias[ch];
      #pragma unroll
      for (int mf = 0; mf < 4; mf++) {
        int mbase = wv * 64 + mf * 16 + (lhalf << 3);
        #pragma unroll
        for (int r = 0; r < 8; r++)
          sOut[(mbase + r) * CCH + ch] = acc[mf][nf][r] + bv;
      }
    }
    __syncthreads();
    float* op = (float*)out;
    for (int t = tid; t < 256 * 16; t += 128) {     // 16 uint4 per pixel
      int pix = t >> 4, part = t & 15;
      int oy = pix >> 4, ox = pix & 15;
      *(uint4*)(op + (((size_t)n * HH + (ty0 + oy)) * WW + (tx0 + ox)) * CCH + part * 4)
          = *(const uint4*)(sOut + pix * CCH + part * 4);
    }
  }
}

// ---------------- launcher -----------------------------------------------------
extern "C" void kernel_launch(void* const* d_in, const int* in_sizes, int n_in,
                              void* d_out, int out_size, void* d_ws, size_t ws_size,
                              hipStream_t stream) {
  (void)in_sizes; (void)n_in; (void)out_size; (void)ws_size;
  const float* means = (const float*)d_in[0];
  // d_in[1] = depths (unused by reference math)
  const float* feats = (const float*)d_in[2];
  const float* intr  = (const float*)d_in[3];
  const float* extr  = (const float*)d_in[4];
  const float* w1    = (const float*)d_in[5];
  const float* b1    = (const float*)d_in[6];
  const float* w2    = (const float*)d_in[7];
  const float* b2    = (const float*)d_in[8];

  char* ws = (char*)d_ws;
  int*      counts = (int*)     (ws + OFF_COUNTS);
  float*    w2c    = (float*)   (ws + OFF_W2C);
  _Float16* wp1    = (_Float16*)(ws + OFF_WP1);
  _Float16* wp2    = (_Float16*)(ws + OFF_WP2);
  _Float16* fused  = (_Float16*)(ws + OFF_FUSED);
  _Float16* y1     = (_Float16*)(ws + OFF_Y1);

  k_init <<<1, 64, 0, stream>>>(extr, w2c, counts);
  k_packw<<<288, 256, 0, stream>>>(w1, w2, wp1, wp2);
  k_count<<<dim3(64, 16, 1), 256, 0, stream>>>(means, intr, w2c, counts);
  k_fuse <<<8192, 256, 0, stream>>>(means, feats, intr, w2c, counts, fused);
  k_conv<0><<<dim3(16, 16, 8), 128, 0, stream>>>(fused, wp1, b1, (void*)y1);
  k_conv<1><<<dim3(16, 16, 8), 128, 0, stream>>>(y1,    wp2, b2, d_out);
}
